// ResidualAttentionBlock_68143951118592
// MI455X (gfx1250) — matrix operand order repaired and checked by reading, compiled, not run on hardware
//
#include <hip/hip_runtime.h>
#include <hip/hip_bf16.h>

// ---------------------------------------------------------------------------
// ResidualAttentionBlock for gfx1250 (MI455X): B=4, S=2048, E=512, H=8, HD=64
// All GEMMs via v_wmma_f32_16x16x32_f16 (f16 in, f32 accumulate).
// GEMMs are 4x4 register-blocked (64x64 tile per wave, 16 independent
// accumulators) so consecutive v_wmma ops are independent (no hazard NOPs)
// and each fragment load feeds 4 WMMAs.
// ---------------------------------------------------------------------------

typedef __attribute__((ext_vector_type(16))) _Float16 v16h;
typedef __attribute__((ext_vector_type(8)))  _Float16 v8h;
typedef __attribute__((ext_vector_type(8)))  float    v8f;

#define WMMA_F16(a, b, c) \
  __builtin_amdgcn_wmma_f32_16x16x32_f16(false, (a), false, (b), (short)0, (c), false, false)

static constexpr int B_  = 4;
static constexpr int S_  = 2048;
static constexpr int E_  = 512;
static constexpr int H_  = 8;
static constexpr int HD_ = 64;
static constexpr int MLP_ = 2048;
static constexpr int ROWS_ = B_ * S_;          // 8192
static constexpr float EPS_ = 1e-5f;

// --------------------------- fragment helpers ------------------------------
// NT GEMM, row-major A[M,K], row-major W[N,K], fragment layouts used here (contiguous k, same for A and B):
//  A-frag (16x32 f16): lane l (lo=l&15, hi=l>>4): M=lo,
//    elems e = A[lo][k0 + hi*16 + e]
//  B-frag (32x16 f16): lane l: N=lo, elems e = W[lo][k0 + hi*16 + e]
//  C/D (16x16 f32): lane l: N=lo, elem r: M = 8*hi + r

__device__ __forceinline__ v16h load_a_frag(const _Float16* __restrict__ arow,
                                            int k0, int hi) {
  v8h lo8 = *(const v8h*)(arow + k0 + hi * 16);
  v8h hi8 = *(const v8h*)(arow + k0 + hi * 16 + 8);
  return __builtin_shufflevector(lo8, hi8, 0, 1, 2, 3, 4, 5, 6, 7,
                                 8, 9, 10, 11, 12, 13, 14, 15);
}

__device__ __forceinline__ v16h load_b_frag(const _Float16* __restrict__ wrow,
                                            int k0, int hi) {
  return *(const v16h*)(wrow + k0 + hi * 16);
}

// 16-lane (half-wave) reductions; wave32, width=16 keeps it in each half.
__device__ __forceinline__ float half16_max(float v) {
#pragma unroll
  for (int m = 1; m < 16; m <<= 1) v = fmaxf(v, __shfl_xor(v, m, 16));
  return v;
}
__device__ __forceinline__ float half16_sum(float v) {
#pragma unroll
  for (int m = 1; m < 16; m <<= 1) v += __shfl_xor(v, m, 16);
  return v;
}

// --------------------------- small kernels ---------------------------------

__global__ void cvt_f32_to_f16(const float* __restrict__ in,
                               _Float16* __restrict__ out, int n) {
  int i = blockIdx.x * blockDim.x + threadIdx.x;
  if (i < n) out[i] = (_Float16)in[i];
}

// One block (256 thr) per row of E=512: LayerNorm -> f16 output.
__global__ void layernorm_to_f16(const float* __restrict__ in,
                                 const float* __restrict__ g,
                                 const float* __restrict__ b,
                                 _Float16* __restrict__ out) {
  __shared__ float s1[256];
  __shared__ float s2[256];
  const int row = blockIdx.x;
  const int t = threadIdx.x;
  const float* x = in + (size_t)row * E_;
  float x0 = x[t], x1 = x[t + 256];
  s1[t] = x0 + x1;
  s2[t] = x0 * x0 + x1 * x1;
  __syncthreads();
#pragma unroll
  for (int st = 128; st > 0; st >>= 1) {
    if (t < st) { s1[t] += s1[t + st]; s2[t] += s2[t + st]; }
    __syncthreads();
  }
  const float mu  = s1[0] * (1.0f / E_);
  const float var = s2[0] * (1.0f / E_) - mu * mu;
  const float rs  = rsqrtf(var + EPS_);
  _Float16* o = out + (size_t)row * E_;
  o[t]       = (_Float16)((x0 - mu) * rs * g[t] + b[t]);
  o[t + 256] = (_Float16)((x1 - mu) * rs * g[t + 256] + b[t + 256]);
}

// Mean over heads: aw[b,q,k] = (1/8) sum_h probs[b,h,q,k].  S*S = 2^22.
__global__ void head_mean(const _Float16* __restrict__ probs,
                          float* __restrict__ aw) {
  size_t idx = (size_t)blockIdx.x * blockDim.x + threadIdx.x;
  size_t bb = idx >> 22;
  size_t r = idx & ((1u << 22) - 1);
  float s = 0.0f;
#pragma unroll
  for (int h = 0; h < H_; ++h)
    s += (float)probs[(((bb * H_) + h) << 22) + r];
  aw[idx] = s * 0.125f;
}

// --------------------------- generic NT GEMM -------------------------------
// C[M,N] = A[M,K] @ W[N,K]^T ; one wave (32 thr) per 64x64 tile,
// 4x4 register blocking (16 independent v8f accumulators).
// MODE 0: outH = f16(acc + bias[n])
// MODE 1: outF = acc + bias[n] + resid[m*N+n]       (fp32)
// MODE 2: outH = f16( gelu_exact(acc + bias[n]) )
template <int MODE>
__global__ void gemm_nt(const _Float16* __restrict__ A,
                        const _Float16* __restrict__ W,
                        const float* __restrict__ bias,
                        const float* __restrict__ resid,
                        float* __restrict__ outF,
                        _Float16* __restrict__ outH,
                        int N, int K) {
  const int lane = threadIdx.x;
  const int lo = lane & 15, hi = lane >> 4;
  const int mbase = blockIdx.x * 64;
  const int nbase = blockIdx.y * 64;

  const _Float16* arow[4];
  const _Float16* wrow[4];
#pragma unroll
  for (int i = 0; i < 4; ++i) {
    arow[i] = A + (size_t)(mbase + i * 16 + lo) * K;
    wrow[i] = W + (size_t)(nbase + i * 16 + lo) * K;
  }

  v8f acc[4][4];
#pragma unroll
  for (int i = 0; i < 4; ++i)
#pragma unroll
    for (int j = 0; j < 4; ++j) acc[i][j] = (v8f){};

  for (int k0 = 0; k0 < K; k0 += 32) {
    v16h af[4];
#pragma unroll
    for (int i = 0; i < 4; ++i) af[i] = load_a_frag(arow[i], k0, hi);
#pragma unroll
    for (int j = 0; j < 4; ++j) {
      v16h bf = load_b_frag(wrow[j], k0, hi);
#pragma unroll
      for (int i = 0; i < 4; ++i) acc[i][j] = WMMA_F16(af[i], bf, acc[i][j]);
    }
  }

#pragma unroll
  for (int j = 0; j < 4; ++j) {
    const int n = nbase + j * 16 + lo;
    const float bn = bias[n];
#pragma unroll
    for (int i = 0; i < 4; ++i) {
#pragma unroll
      for (int r = 0; r < 8; ++r) {
        const int m = mbase + i * 16 + 8 * hi + r;
        float v = acc[i][j][r] + bn;
        if (MODE == 1) {
          v += resid[(size_t)m * N + n];
          outF[(size_t)m * N + n] = v;
        } else if (MODE == 2) {
          v = 0.5f * v * (1.0f + erff(v * 0.70710678118654752f));
          outH[(size_t)m * N + n] = (_Float16)v;
        } else {
          outH[(size_t)m * N + n] = (_Float16)v;
        }
      }
    }
  }
}

// --------------------- fused scores + online softmax -----------------------
// grid: (S/16 q-tiles, B*H). block: 128 thr = 4 waves; wave w owns key tiles
// w, w+4, ... Two passes over K-tiles with online (max,sum); merge across
// waves in LDS; pass 2 writes normalized f16 probs [B,H,S,S].
__global__ void attn_scores_softmax(const _Float16* __restrict__ qkv,
                                    _Float16* __restrict__ probs) {
  __shared__ float smax[4][16];
  __shared__ float ssum[4][16];
  const int wave = threadIdx.x >> 5;
  const int lane = threadIdx.x & 31;
  const int lo = lane & 15, hi = lane >> 4;
  const int bh = blockIdx.y;
  const int b = bh >> 3, h = bh & 7;
  const int mbase = blockIdx.x * 16;

  // q rows for this tile, K=0..63 (two 32-wide frags), loaded once.
  const _Float16* qrow = qkv + ((size_t)(b * S_ + mbase + lo) * (3 * E_) + h * HD_);
  const v16h a0 = load_a_frag(qrow, 0, hi);
  const v16h a1 = load_a_frag(qrow, 32, hi);

  float rmax[8], rsum[8];
#pragma unroll
  for (int r = 0; r < 8; ++r) { rmax[r] = -3.4e38f; rsum[r] = 0.0f; }

  // ---- pass 1: online max/sum over this wave's key tiles ----
  for (int nt = wave; nt < S_ / 16; nt += 4) {
    const _Float16* krow =
        qkv + ((size_t)(b * S_ + nt * 16 + lo) * (3 * E_) + E_ + h * HD_);
    v16h b0 = load_b_frag(krow, 0, hi);
    v16h b1 = load_b_frag(krow, 32, hi);
    v8f c = {};
    c = WMMA_F16(a0, b0, c);
    c = WMMA_F16(a1, b1, c);
#pragma unroll
    for (int r = 0; r < 8; ++r) {
      float s = c[r] * 0.125f;              // 1/sqrt(64)
      float tmax = half16_max(s);
      float nm = fmaxf(rmax[r], tmax);
      float p = __expf(s - nm);
      float tsum = half16_sum(p);
      rsum[r] = rsum[r] * __expf(rmax[r] - nm) + tsum;
      rmax[r] = nm;
    }
  }
  // ---- merge stats across the 4 waves ----
  if (lo == 0) {
#pragma unroll
    for (int r = 0; r < 8; ++r) {
      smax[wave][8 * hi + r] = rmax[r];
      ssum[wave][8 * hi + r] = rsum[r];
    }
  }
  __syncthreads();
  float gmax[8], ginv[8];
#pragma unroll
  for (int r = 0; r < 8; ++r) {
    const int row = 8 * hi + r;
    float gm = smax[0][row];
#pragma unroll
    for (int w = 1; w < 4; ++w) gm = fmaxf(gm, smax[w][row]);
    float gs = 0.0f;
#pragma unroll
    for (int w = 0; w < 4; ++w) gs += ssum[w][row] * __expf(smax[w][row] - gm);
    gmax[r] = gm;
    ginv[r] = 1.0f / gs;
  }
  // ---- pass 2: recompute tiles, emit normalized f16 probs ----
  for (int nt = wave; nt < S_ / 16; nt += 4) {
    const _Float16* krow =
        qkv + ((size_t)(b * S_ + nt * 16 + lo) * (3 * E_) + E_ + h * HD_);
    v16h b0 = load_b_frag(krow, 0, hi);
    v16h b1 = load_b_frag(krow, 32, hi);
    v8f c = {};
    c = WMMA_F16(a0, b0, c);
    c = WMMA_F16(a1, b1, c);
#pragma unroll
    for (int r = 0; r < 8; ++r) {
      float p = __expf(c[r] * 0.125f - gmax[r]) * ginv[r];
      probs[((size_t)bh * S_ + mbase + 8 * hi + r) * S_ + nt * 16 + lo] =
          (_Float16)p;
    }
  }
}

// --------------------------- ctx = probs @ V -------------------------------
// grid: (S/64 m-tiles, B*H). block 128 thr = 4 waves; wave w owns n-cols
// [w*16, w*16+16) of HD=64 and 64 rows (4 m sub-tiles). The gathered V
// B-frag (16 strided u16 loads) is amortized over 4 WMMAs.
__global__ void attn_ctx(const _Float16* __restrict__ probs,
                         const _Float16* __restrict__ qkv,
                         _Float16* __restrict__ ctx) {
  const int wave = threadIdx.x >> 5;
  const int lane = threadIdx.x & 31;
  const int lo = lane & 15, hi = lane >> 4;
  const int bh = blockIdx.y;
  const int b = bh >> 3, h = bh & 7;
  const int mbase = blockIdx.x * 64;
  const int nb = wave * 16;

  const _Float16* prow[4];
#pragma unroll
  for (int i = 0; i < 4; ++i)
    prow[i] = probs + ((size_t)bh * S_ + mbase + i * 16 + lo) * S_;

  v8f acc[4];
#pragma unroll
  for (int i = 0; i < 4; ++i) acc[i] = (v8f){};

  for (int k0 = 0; k0 < S_; k0 += 32) {
    // B-frag: elem e -> V[b, k0 + hi*16 + e, h, nb + lo]
    const _Float16* vcol = qkv + ((size_t)(b * S_ + k0 + hi * 16) * (3 * E_) +
                                  2 * E_ + h * HD_ + nb + lo);
    v16h bf;
#pragma unroll
    for (int e = 0; e < 16; ++e) bf[e] = vcol[(size_t)e * (3 * E_)];
#pragma unroll
    for (int i = 0; i < 4; ++i) {
      v16h af = load_a_frag(prow[i], k0, hi);
      acc[i] = WMMA_F16(af, bf, acc[i]);
    }
  }
#pragma unroll
  for (int i = 0; i < 4; ++i) {
#pragma unroll
    for (int r = 0; r < 8; ++r) {
      ctx[((size_t)(b * S_ + mbase + i * 16 + 8 * hi + r)) * E_ + h * HD_ +
          nb + lo] = (_Float16)acc[i][r];
    }
  }
}

// ----------------------------- launcher ------------------------------------
extern "C" void kernel_launch(void* const* d_in, const int* in_sizes, int n_in,
                              void* d_out, int out_size, void* d_ws, size_t ws_size,
                              hipStream_t stream) {
  (void)in_sizes; (void)n_in; (void)out_size; (void)ws_size;
  const float* query     = (const float*)d_in[0];
  const float* ln1_g     = (const float*)d_in[1];
  const float* ln1_b     = (const float*)d_in[2];
  const float* in_proj_w = (const float*)d_in[3];
  const float* in_proj_b = (const float*)d_in[4];
  const float* out_w     = (const float*)d_in[5];
  const float* out_b     = (const float*)d_in[6];
  const float* ln2_g     = (const float*)d_in[7];
  const float* ln2_b     = (const float*)d_in[8];
  const float* w1        = (const float*)d_in[9];
  const float* b1        = (const float*)d_in[10];
  const float* w2        = (const float*)d_in[11];
  const float* b2        = (const float*)d_in[12];

  float* out_x  = (float*)d_out;                       // [B,S,E]
  float* out_aw = out_x + (size_t)B_ * S_ * E_;        // [B,S,S]

  // workspace carve-up (all 256B aligned)
  char* p = (char*)d_ws;
  auto alloc = [&](size_t bytes) {
    void* q = (void*)p;
    p += (bytes + 255) & ~(size_t)255;
    return q;
  };
  _Float16* wq_h   = (_Float16*)alloc((size_t)3 * E_ * E_ * 2);      // [3E,E]
  _Float16* wo_h   = (_Float16*)alloc((size_t)E_ * E_ * 2);          // [E,E]
  _Float16* w1_h   = (_Float16*)alloc((size_t)MLP_ * E_ * 2);        // [MLP,E]
  _Float16* w2_h   = (_Float16*)alloc((size_t)E_ * MLP_ * 2);        // [E,MLP]
  _Float16* xq_h   = (_Float16*)alloc((size_t)ROWS_ * E_ * 2);       // ln1 out
  _Float16* qkv_h  = (_Float16*)alloc((size_t)ROWS_ * 3 * E_ * 2);   // qkv
  _Float16* ctx_h  = (_Float16*)alloc((size_t)ROWS_ * E_ * 2);       // attn ctx
  _Float16* h_h    = (_Float16*)alloc((size_t)ROWS_ * E_ * 2);       // ln2 out
  _Float16* m_h    = (_Float16*)alloc((size_t)ROWS_ * MLP_ * 2);     // gelu out
  float*    x_res  = (float*)alloc((size_t)ROWS_ * E_ * 4);          // x + attn
  _Float16* probs_h = (_Float16*)alloc((size_t)B_ * H_ * S_ * S_ * 2);

  // 1) weights -> f16
  cvt_f32_to_f16<<<(3 * E_ * E_ + 255) / 256, 256, 0, stream>>>(in_proj_w, wq_h, 3 * E_ * E_);
  cvt_f32_to_f16<<<(E_ * E_ + 255) / 256, 256, 0, stream>>>(out_w, wo_h, E_ * E_);
  cvt_f32_to_f16<<<(MLP_ * E_ + 255) / 256, 256, 0, stream>>>(w1, w1_h, MLP_ * E_);
  cvt_f32_to_f16<<<(E_ * MLP_ + 255) / 256, 256, 0, stream>>>(w2, w2_h, E_ * MLP_);

  // 2) ln1(query) -> xq_h
  layernorm_to_f16<<<ROWS_, 256, 0, stream>>>(query, ln1_g, ln1_b, xq_h);

  // 3) qkv = xq @ in_proj_w^T + b       (M=8192, N=1536, K=512)
  gemm_nt<0><<<dim3(ROWS_ / 64, 3 * E_ / 64), 32, 0, stream>>>(
      xq_h, wq_h, in_proj_b, nullptr, nullptr, qkv_h, 3 * E_, E_);

  // 4) scores + softmax -> probs_h [B,H,S,S]
  attn_scores_softmax<<<dim3(S_ / 16, B_ * H_), 128, 0, stream>>>(qkv_h, probs_h);

  // 5) attn_weight = mean over heads
  head_mean<<<(int)(((size_t)B_ * S_ * S_) / 256), 256, 0, stream>>>(probs_h, out_aw);

  // 6) ctx = probs @ v -> ctx_h [B,S,E]
  attn_ctx<<<dim3(S_ / 64, B_ * H_), 128, 0, stream>>>(probs_h, qkv_h, ctx_h);

  // 7) x_res = query + ctx @ out_w^T + out_b   (M=8192, N=512, K=512)
  gemm_nt<1><<<dim3(ROWS_ / 64, E_ / 64), 32, 0, stream>>>(
      ctx_h, wo_h, out_b, query, x_res, nullptr, E_, E_);

  // 8) ln2(x_res) -> h_h
  layernorm_to_f16<<<ROWS_, 256, 0, stream>>>(x_res, ln2_g, ln2_b, h_h);

  // 9) m = gelu(h @ w1^T + b1)          (M=8192, N=2048, K=512)
  gemm_nt<2><<<dim3(ROWS_ / 64, MLP_ / 64), 32, 0, stream>>>(
      h_h, w1_h, b1, nullptr, nullptr, m_h, MLP_, E_);

  // 10) out_x = x_res + m @ w2^T + b2   (M=8192, N=512, K=2048)
  gemm_nt<1><<<dim3(ROWS_ / 64, E_ / 64), 32, 0, stream>>>(
      m_h, w2_h, b2, x_res, out_x, nullptr, E_, MLP_);
}
